// GATNet_10582799418097
// MI455X (gfx1250) — matrix operand order repaired and checked by reading, not compiled, not run on hardware
//
#include <hip/hip_runtime.h>
#include <hip/hip_bf16.h>
#include <math.h>

// ---------------- problem constants (from reference) ----------------
#define N_NODES   50000
#define E_EDGES   800000
#define TE_EDGES  (E_EDGES + N_NODES)   // edges + self loops = 850000
#define NEG_SLOPE 0.2f
#define ROW_TILES (N_NODES / 16)        // 3125 (exact)

typedef float    v2f  __attribute__((ext_vector_type(2)));
typedef float    v8f  __attribute__((ext_vector_type(8)));
typedef _Float16 v16h __attribute__((ext_vector_type(16)));

// Prefer exact fp32 WMMA (16x16x4); fall back to f16 WMMA if not declared.
#if defined(__HIP_DEVICE_COMPILE__)
# if __has_builtin(__builtin_amdgcn_wmma_f32_16x16x4_f32)
#  define USE_WMMA_F32 1
# elif __has_builtin(__builtin_amdgcn_wmma_f32_16x16x32_f16)
#  define USE_WMMA_F16 1
# endif
#endif

// ---------------- device helpers ----------------
__device__ __forceinline__ float leaky(float x) { return x > 0.f ? x : NEG_SLOPE * x; }

// order-preserving float<->uint key for atomic max (init key = 0 == below all reals)
__device__ __forceinline__ unsigned f32key(float f) {
  int i = __float_as_int(f);
  return (i >= 0) ? ((unsigned)i | 0x80000000u) : ~(unsigned)i;
}
__device__ __forceinline__ float keyf32(unsigned k) {
  int i = (k & 0x80000000u) ? (int)(k & 0x7fffffffu) : ~(int)k;
  return __int_as_float(i);
}

// Native fp32 atomic add (no CAS loop); RMW resolves at device scope (L2).
__device__ __forceinline__ void atomic_add_f32(float* p, float v) {
  asm volatile("global_atomic_add_f32 %0, %1, off scope:SCOPE_DEV"
               :: "v"(p), "v"(v) : "memory");
}

// Async global->LDS DMA (ASYNCcnt-tracked), per CDNA5 ISA.
__device__ __forceinline__ void async_copy_b128(unsigned lds_off, const void* gaddr) {
  asm volatile("global_load_async_to_lds_b128 %0, %1, off"
               :: "v"(lds_off), "v"(gaddr) : "memory");
}
__device__ __forceinline__ void wait_async0() {
  asm volatile("s_wait_asynccnt 0x0" ::: "memory");
}

// edge_index is int64 per the reference (shape [2,E], row0=src, row1=dst)
__device__ __forceinline__ void edge_sd(const long long* __restrict__ ei, int e,
                                        int& s, int& d) {
  if (e < E_EDGES) { s = (int)ei[e]; d = (int)ei[E_EDGES + e]; }
  else             { s = d = e - E_EDGES; }              // self loop
}

// ---------------- kernels ----------------
__global__ void zero_u32_kernel(unsigned* __restrict__ p, int n) {
  int t = blockIdx.x * blockDim.x + threadIdx.x;
  if (t < n) p[t] = 0u;
}

// C[N_NODES x NC] = A[N_NODES x K] * B[K x NC].
// Block = 256 threads = 8 waves. colTiles = NC/16, R = 8/colTiles row tiles
// per block. The block's A rows (contiguous R*16*K floats) are DMA'd to LDS
// once via async b128 copies; each wave then computes one 16x16 WMMA tile
// reading A-fragments from LDS (ds_load_b64) and B-fragments from global (L2).
__global__ void wmma_gemm_lds_kernel(const float* __restrict__ A, const float* __restrict__ B,
                                     float* __restrict__ C, int K, int NC) {
  extern __shared__ float ldsA[];               // R*16*K floats
  const int tid      = threadIdx.x;
  const int lane     = tid & 31;
  const int wave     = tid >> 5;
  const int colTiles = NC >> 4;                 // 8 (gemm1) or 1 (gemm2)
  const int R        = 8 / colTiles;            // row tiles per block
  const int blockRow0 = blockIdx.x * R * 16;

  // ---- stage A rows into LDS with async DMA ----
  {
    const unsigned ldsbase = (unsigned)(uintptr_t)ldsA;  // low 32 bits = LDS offset
    const float*   gbase   = A + (size_t)blockRow0 * K;
    const int      nvec    = (R * 16 * K) >> 2;          // float4 chunks
    for (int i = tid; i < nvec; i += 256)
      async_copy_b128(ldsbase + (unsigned)i * 16u, gbase + (size_t)i * 4);
    wait_async0();
  }
  __syncthreads();

  const int rtl  = wave / colTiles;             // local row tile
  const int ct   = wave - rtl * colTiles;       // col tile
  const int r0   = blockRow0 + rtl * 16;        // global row base
  const int c0   = ct << 4;
  const int mn   = lane & 15;                   // M for A-frag, N for B/C/D
  const int half = lane >> 4;

  v8f acc = {};
  const float* arow = ldsA + (size_t)(rtl * 16 + mn) * K;
#if defined(USE_WMMA_F32)
  // A 16x4 f32 frag: vgpr0 = K=half*2, vgpr1 = K=half*2+1 (ISA 7.12.2)
  for (int k = 0; k < K; k += 4) {
    int kk = k + half * 2;
    v2f a, b;
    a.x = arow[kk];                             // ds_load pair from LDS
    a.y = arow[kk + 1];
    b.x = B[(size_t)kk       * NC + c0 + mn];
    b.y = B[(size_t)(kk + 1) * NC + c0 + mn];
    acc = __builtin_amdgcn_wmma_f32_16x16x4_f32(false, a, false, b,
                                                (short)0, acc, false, false);
  }
#elif defined(USE_WMMA_F16)
  for (int k0 = 0; k0 < K; k0 += 32) {
    v16h a, b;
#pragma unroll
    for (int e2 = 0; e2 < 16; ++e2) {
      int v  = e2 >> 1, hh = e2 & 1;
      int ka = ((v >> 2) << 4) + (half << 3) + ((v & 3) << 1) + hh;
      a[e2] = (_Float16)arow[k0 + ka];
      int kb = ((e2 >> 3) << 4) + (half << 3) + (e2 & 7);
      b[e2] = (_Float16)B[(size_t)(k0 + kb) * NC + c0 + mn];
    }
    acc = __builtin_amdgcn_wmma_f32_16x16x32_f16(false, a, false, b,
                                                 (short)0, acc, false, false);
  }
#else
  (void)arow; // host-compile placeholder; device always takes a WMMA path
#endif
  // C/D: vgpr r -> M = r + 8*half, N = mn  (guard tail rows of last block)
  float* crow = C + c0 + mn;
#pragma unroll
  for (int r = 0; r < 8; ++r) {
    int m = r0 + r + (half << 3);
    if (m < N_NODES) crow[(size_t)m * NC] = acc[r];
  }
}

// alpha_s/alpha_d for layer 1: one node per block, one head per wave32.
__global__ void alpha1_kernel(const float* __restrict__ xp, const float* __restrict__ asrc,
                              const float* __restrict__ adst, float* __restrict__ as,
                              float* __restrict__ ad) {
  int n = blockIdx.x;
  int t = threadIdx.x;        // 0..127  (h = t>>5, c = t&31)
  int h = t >> 5, c = t & 31;
  float v = xp[(size_t)n * 128 + t];
  float s = v * asrc[t];
  float d = v * adst[t];
  for (int off = 16; off > 0; off >>= 1) {   // wave32 tree reduce
    s += __shfl_down(s, off, 32);
    d += __shfl_down(d, off, 32);
  }
  if (c == 0) { as[n * 4 + h] = s; ad[n * 4 + h] = d; }
}

// alpha_s/alpha_d for layer 2 (1 head, 16 channels): one node per thread.
__global__ void alpha2_kernel(const float* __restrict__ xp, const float* __restrict__ asrc,
                              const float* __restrict__ adst, float* __restrict__ as,
                              float* __restrict__ ad) {
  int n = blockIdx.x * blockDim.x + threadIdx.x;
  if (n >= N_NODES) return;
  float s = 0.f, d = 0.f;
  for (int c = 0; c < 16; ++c) {
    float v = xp[(size_t)n * 16 + c];
    s += v * asrc[c];
    d += v * adst[c];
  }
  as[n] = s; ad[n] = d;
}

__global__ void edge_max_kernel(const long long* __restrict__ ei, const float* __restrict__ as,
                                const float* __restrict__ ad, unsigned* __restrict__ m,
                                int logH, int total) {
  int t = blockIdx.x * blockDim.x + threadIdx.x;
  if (t >= total) return;
  int H = 1 << logH;
  int e = t >> logH, hh = t & (H - 1);
  int s, d; edge_sd(ei, e, s, d);
  float v = leaky(as[s * H + hh] + ad[d * H + hh]);
  atomicMax(&m[d * H + hh], f32key(v));
}

__global__ void edge_den_kernel(const long long* __restrict__ ei, const float* __restrict__ as,
                                const float* __restrict__ ad, const unsigned* __restrict__ m,
                                float* __restrict__ den, int logH, int total) {
  int t = blockIdx.x * blockDim.x + threadIdx.x;
  if (t >= total) return;
  int H = 1 << logH;
  int e = t >> logH, hh = t & (H - 1);
  int s, d; edge_sd(ei, e, s, d);
  float v  = leaky(as[s * H + hh] + ad[d * H + hh]);
  float ex = expf(v - keyf32(m[d * H + hh]));
  atomic_add_f32(&den[d * H + hh], ex);
}

// out[dst, col] += xp[src, col] * softmax_alpha(e, head(col))
__global__ void edge_scatter_kernel(const long long* __restrict__ ei, const float* __restrict__ as,
                                    const float* __restrict__ ad, const unsigned* __restrict__ m,
                                    const float* __restrict__ den, const float* __restrict__ xp,
                                    float* __restrict__ out, int logH, int logC, int total) {
  int t = blockIdx.x * blockDim.x + threadIdx.x;
  if (t >= total) return;
  int logF = logH + logC;
  int F = 1 << logF, H = 1 << logH;
  int e = t >> logF, col = t & (F - 1), hh = col >> logC;
  int s, d; edge_sd(ei, e, s, d);
  float v     = leaky(as[s * H + hh] + ad[d * H + hh]);
  float alpha = expf(v - keyf32(m[d * H + hh])) / den[d * H + hh];
  atomic_add_f32(&out[(size_t)d * F + col], xp[(size_t)s * F + col] * alpha);
}

__global__ void elu_bias_kernel(float* __restrict__ h, const float* __restrict__ b, int total) {
  int t = blockIdx.x * blockDim.x + threadIdx.x;
  if (t >= total) return;
  float v = h[t] + b[t & 127];
  h[t] = v > 0.f ? v : expm1f(v);
}

__global__ void logsoftmax_kernel(const float* __restrict__ acc, const float* __restrict__ b,
                                  float* __restrict__ out) {
  int n = blockIdx.x * blockDim.x + threadIdx.x;
  if (n >= N_NODES) return;
  float v[16]; float mx = -INFINITY;
#pragma unroll
  for (int c = 0; c < 16; ++c) { v[c] = acc[(size_t)n * 16 + c] + b[c]; mx = fmaxf(mx, v[c]); }
  float sum = 0.f;
#pragma unroll
  for (int c = 0; c < 16; ++c) sum += expf(v[c] - mx);
  float lse = mx + logf(sum);
#pragma unroll
  for (int c = 0; c < 16; ++c) out[(size_t)n * 16 + c] = v[c] - lse;
}

// ---------------- host launch ----------------
extern "C" void kernel_launch(void* const* d_in, const int* in_sizes, int n_in,
                              void* d_out, int out_size, void* d_ws, size_t ws_size,
                              hipStream_t stream) {
  const float*     x    = (const float*)d_in[0];
  const long long* ei   = (const long long*)d_in[1];   // int64 per reference
  const float*     W1   = (const float*)d_in[2];
  const float*     a_s1 = (const float*)d_in[3];
  const float*     a_d1 = (const float*)d_in[4];
  const float*     b1   = (const float*)d_in[5];
  const float*     W2   = (const float*)d_in[6];
  const float*     a_s2 = (const float*)d_in[7];
  const float*     a_d2 = (const float*)d_in[8];
  const float*     b2   = (const float*)d_in[9];
  float*           out  = (float*)d_out;

  // workspace layout (~62 MB; entire working set fits in 192 MB L2)
  float*    xp1  = (float*)d_ws;                               // N*128
  float*    as1  = xp1  + (size_t)N_NODES * 128;               // N*4
  float*    ad1  = as1  + (size_t)N_NODES * 4;                 // N*4
  unsigned* m1   = (unsigned*)(ad1 + (size_t)N_NODES * 4);     // N*4 keys
  float*    den1 = (float*)(m1 + (size_t)N_NODES * 4);         // N*4
  float*    h1   = den1 + (size_t)N_NODES * 4;                 // N*128 accum -> ELU(h)
  float*    xp2  = h1   + (size_t)N_NODES * 128;               // N*16
  float*    as2  = xp2  + (size_t)N_NODES * 16;                // N
  float*    ad2  = as2  + N_NODES;                             // N
  unsigned* m2   = (unsigned*)(ad2 + N_NODES);                 // N keys
  float*    den2 = (float*)(m2 + N_NODES);                     // N
  float*    acc2 = den2 + N_NODES;                             // N*16

  const int B = 256;

  // zero [m1 .. acc2 end): 4+4+128+16+1+1+1+1+16 = 172 words per node
  int zn = 172 * N_NODES;
  zero_u32_kernel<<<(zn + B - 1) / B, B, 0, stream>>>(m1, zn);

  // layer 1: xp1 = x @ W1.  R=1 row tile, 8 col tiles per block; 8 KB LDS.
  wmma_gemm_lds_kernel<<<ROW_TILES, 256, 1 * 16 * 128 * 4, stream>>>(x, W1, xp1, 128, 128);

  alpha1_kernel<<<N_NODES, 128, 0, stream>>>(xp1, a_s1, a_d1, as1, ad1);

  int t1 = TE_EDGES * 4;                       // (edge, head)
  edge_max_kernel<<<(t1 + B - 1) / B, B, 0, stream>>>(ei, as1, ad1, m1, 2, t1);
  edge_den_kernel<<<(t1 + B - 1) / B, B, 0, stream>>>(ei, as1, ad1, m1, den1, 2, t1);
  int t2 = TE_EDGES * 128;                     // (edge, column)
  edge_scatter_kernel<<<(t2 + B - 1) / B, B, 0, stream>>>(ei, as1, ad1, m1, den1,
                                                          xp1, h1, 2, 5, t2);

  int t3 = N_NODES * 128;
  elu_bias_kernel<<<(t3 + B - 1) / B, B, 0, stream>>>(h1, b1, t3);

  // layer 2: xp2 = h1 @ W2.  R=8 row tiles, 1 col tile per block; 64 KB LDS.
  // (tail block over-reads past h1 into our own workspace -> safe; stores guarded)
  int g2 = (ROW_TILES + 7) / 8;                // 391
  wmma_gemm_lds_kernel<<<g2, 256, 8 * 16 * 128 * 4, stream>>>(h1, W2, xp2, 128, 16);

  alpha2_kernel<<<(N_NODES + B - 1) / B, B, 0, stream>>>(xp2, a_s2, a_d2, as2, ad2);

  int t4 = TE_EDGES;
  edge_max_kernel<<<(t4 + B - 1) / B, B, 0, stream>>>(ei, as2, ad2, m2, 0, t4);
  edge_den_kernel<<<(t4 + B - 1) / B, B, 0, stream>>>(ei, as2, ad2, m2, den2, 0, t4);
  int t5 = TE_EDGES * 16;
  edge_scatter_kernel<<<(t5 + B - 1) / B, B, 0, stream>>>(ei, as2, ad2, m2, den2,
                                                          xp2, acc2, 0, 4, t5);

  logsoftmax_kernel<<<(N_NODES + B - 1) / B, B, 0, stream>>>(acc2, b2, out);
}
